// GenePropCA_7344394076676
// MI455X (gfx1250) — compile-verified, matrix operand order checked by reading, not run on hardware
//
#include <hip/hip_runtime.h>
#include <hip/hip_bf16.h>
#include <math.h>

#define B_   16
#define C_   23
#define H_   256
#define W_   256
#define HW_  (H_*W_)
#define G_   3
#define HID_ 96
#define FEAT_ 92           // 4*C
#define DT_  0.1f

typedef __attribute__((ext_vector_type(16))) _Float16 v16h;
typedef __attribute__((ext_vector_type(8)))  _Float16 v8h;
typedef __attribute__((ext_vector_type(8)))  float    v8f;

// ---------------------------------------------------------------------------
// Kernel A: oscillator reaction step. Produces na, nb (workspace), new_d and
// copies the 17 unchanged channels into the output x buffer.
// ---------------------------------------------------------------------------
__global__ __launch_bounds__(256) void geneca_osc1(
    const float* __restrict__ x,
    const float* __restrict__ sw, const float* __restrict__ sb,
    const float* __restrict__ pAl, const float* __restrict__ pBe,
    const float* __restrict__ pOm, const float* __restrict__ pK,
    const float* __restrict__ pKp,
    float* __restrict__ xg, float* __restrict__ na, float* __restrict__ nb)
{
  int idx = blockIdx.x * blockDim.x + threadIdx.x;
  if (idx >= B_ * HW_) return;
  int w  = idx & (W_ - 1);
  int h  = (idx >> 8) & (H_ - 1);
  int bi = idx >> 16;

  auto L = [&](int c, int hh, int ww) -> float {
    return x[((size_t)(bi * C_ + c) * H_ + hh) * W_ + ww];
  };

  float nA[3][3], nv[3][3], nw[3][3], nd[3][3];
  float ero = 3.0e38f;
  #pragma unroll
  for (int dy = -1; dy <= 1; ++dy) {
    int hh = h + dy;
    bool hin = (unsigned)hh < (unsigned)H_;
    #pragma unroll
    for (int dx = -1; dx <= 1; ++dx) {
      int ww = w + dx;
      bool in = hin && ((unsigned)ww < (unsigned)W_);
      float va = in ? L(3, hh, ww) : 0.0f;
      nA[dy + 1][dx + 1] = va;
      if (in) ero = fminf(ero, va);   // maxpool3(-x) with -inf pad == min over valid
      nv[dy + 1][dx + 1] = in ? L(16, hh, ww) : 0.0f;
      nw[dy + 1][dx + 1] = in ? L(17, hh, ww) : 0.0f;
      nd[dy + 1][dx + 1] = in ? L(18, hh, ww) : 0.0f;
    }
  }
  auto lapk = [&](const float (&n)[3][3]) -> float {
    return (n[0][0] + n[0][2] + n[2][0] + n[2][2]
            + 2.0f * (n[0][1] + n[1][0] + n[1][2] + n[2][1])
            - 12.0f * n[1][1]) * (1.0f / 12.0f);
  };

  float alpha = nA[1][1];
  float edges = alpha - ero;
  float lapA  = lapk(nA);
  float h4 = L(4, h, w), h5 = L(5, h, w);
  float I0 = sw[0]*alpha + sw[1]*edges + sw[2]*lapA + sw[3]*h4 + sw[4]*h5 + sb[0];
  float I1 = sw[5]*alpha + sw[6]*edges + sw[7]*lapA + sw[8]*h4 + sw[9]*h5 + sb[1];
  float I2 = sw[10]*alpha + sw[11]*edges + sw[12]*lapA + sw[13]*h4 + sw[14]*h5 + sb[2];

  float a = nv[1][1], b = nw[1][1], d = nd[1][1];
  float al = *pAl, be = *pBe, om = *pOm, Kp = *pK, kp = *pKp;
  float va = a + DT_ * (-al * a + om * b + Kp * lapk(nv) + I0);
  float vb = b + DT_ * (-al * b - om * a + Kp * lapk(nw) + I1);
  float vd = d + DT_ * (-be * d + kp * lapk(nd) + I2);

  na[idx] = va;
  nb[idx] = vb;
  xg[((size_t)(bi * C_ + 18) * H_ + h) * W_ + w] = vd;

  // copy unchanged channels 0..15 and 19
  #pragma unroll
  for (int c = 0; c < 16; ++c)
    xg[((size_t)(bi * C_ + c) * H_ + h) * W_ + w] = L(c, h, w);
  xg[((size_t)(bi * C_ + 19) * H_ + h) * W_ + w] = L(19, h, w);
}

// ---------------------------------------------------------------------------
// Kernel B: 5x5 averaging + Kuramoto normalization; writes final ch16/17,
// phase and amplitude outputs.
// ---------------------------------------------------------------------------
__global__ __launch_bounds__(256) void geneca_osc2(
    const float* __restrict__ na, const float* __restrict__ nb,
    float* __restrict__ xg, float* __restrict__ phase, float* __restrict__ amp)
{
  int idx = blockIdx.x * blockDim.x + threadIdx.x;
  if (idx >= B_ * HW_) return;
  int w  = idx & (W_ - 1);
  int h  = (idx >> 8) & (H_ - 1);
  int bi = idx >> 16;

  float sa = 0.0f, sb = 0.0f;
  #pragma unroll
  for (int dy = -2; dy <= 2; ++dy) {
    int hh = h + dy;
    if ((unsigned)hh >= (unsigned)H_) continue;
    #pragma unroll
    for (int dx = -2; dx <= 2; ++dx) {
      int ww = w + dx;
      if ((unsigned)ww >= (unsigned)W_) continue;
      size_t o = (size_t)bi * HW_ + hh * W_ + ww;
      sa += na[o];
      sb += nb[o];
    }
  }
  float aa = sa * (1.0f / 25.0f), bb = sb * (1.0f / 25.0f);
  float rho = sqrtf(aa * aa + bb * bb + 1e-6f);
  aa /= rho; bb /= rho;
  float va = na[idx], vb = nb[idx];
  float fa = va + DT_ * (aa - va);
  float fb = vb + DT_ * (bb - vb);

  xg[((size_t)(bi * C_ + 16) * H_ + h) * W_ + w] = fa;
  xg[((size_t)(bi * C_ + 17) * H_ + h) * W_ + w] = fb;
  size_t p = (size_t)bi * HW_ + h * W_ + w;
  phase[p] = sqrtf(fa * fa + fb * fb + 1e-6f);
  amp[p]   = atan2f(fb, fa);
}

// ---------------------------------------------------------------------------
// Kernel C: per-pixel 92-feature assembly (wrap 3x3 convs; mod channels 20..22
// recomputed on the fly) + 92->96->3 MLP via v_wmma_f32_16x16x32_f16,
// masked gene update into channels 20..22.
// Block = 128 threads = 4 waves; each wave handles 16 consecutive pixels.
// ---------------------------------------------------------------------------
__global__ __launch_bounds__(128) void geneca_mlp(
    const float* __restrict__ xin,    // original input x (channel 3 for life mask)
    float* xg,                        // output x buffer: read ch0..19, write ch20..22
    const float* __restrict__ noise,
    const float* __restrict__ w1w, const float* __restrict__ w1b,
    const float* __restrict__ w2w,
    const float* __restrict__ modw, const float* __restrict__ modb)
{
  // A-operand tiles: [mt*3+k][lane][e], pre-swizzled to the 16-bit A layout.
  __shared__ __align__(32) _Float16 w1L[18][32][16];
  __shared__ float b1L[HID_];
  __shared__ float w2L[G_][HID_];
  __shared__ __align__(16) _Float16 featL[4][16][HID_];   // [wave][pixel][row], rows 92..95 = 0

  const int tid  = threadIdx.x;
  const int lane = tid & 31;
  const int wv   = tid >> 5;
  const int hwb  = lane >> 4;      // half-wave
  const int p    = lane & 15;      // pixel column within wave tile

  // ---- stage weights (K-map: e<8 -> e + 8*hw ; e>=8 -> e + 8 + 8*hw) ----
  for (int i = tid; i < 18 * 32 * 16; i += 128) {
    int e  = i & 15;
    int ln = (i >> 4) & 31;
    int tk = i >> 9;
    int mt = tk / 3, k = tk % 3;
    int o  = mt * 16 + (ln & 15);
    int hb = ln >> 4;
    int K  = (e < 8) ? (e + hb * 8) : (e + 8 + hb * 8);
    int ii = 32 * k + K;
    float v = (ii < FEAT_) ? w1w[o * FEAT_ + ii] : 0.0f;
    ((_Float16*)w1L)[i] = (_Float16)v;
  }
  for (int i = tid; i < HID_; i += 128)      b1L[i] = w1b[i];
  for (int i = tid; i < G_ * HID_; i += 128) ((float*)w2L)[i] = w2w[i];
  for (int i = tid; i < 4 * 16 * HID_; i += 128) ((_Float16*)featL)[i] = (_Float16)0.0f;
  __syncthreads();

  // ---- pixel group ----
  const int blk  = blockIdx.x;            // B*H*4 blocks
  const int wseg = blk & 3;
  const int h    = (blk >> 2) & (H_ - 1);
  const int bi   = blk >> 10;
  const int wcol = wseg * 64 + wv * 16 + p;

  float mw9[9], mb3[3];
  #pragma unroll
  for (int j = 0; j < 9; ++j) mw9[j] = modw[j];
  #pragma unroll
  for (int j = 0; j < 3; ++j) mb3[j] = modb[j];

  auto getv = [&](int c, int hh, int ww) -> float {
    if (c < 20)
      return xg[((size_t)(bi * C_ + c) * H_ + hh) * W_ + ww];
    int j = c - 20;   // mod channels are linear in updated (a,b,d)
    size_t o = ((size_t)bi * C_) * HW_ + (size_t)hh * W_ + ww;
    float fa = xg[o + (size_t)16 * HW_];
    float fb = xg[o + (size_t)17 * HW_];
    float fd = xg[o + (size_t)18 * HW_];
    return mw9[j * 3 + 0] * fa + mw9[j * 3 + 1] * fb + mw9[j * 3 + 2] * fd + mb3[j];
  };

  // ---- features: wrap-padded 3x3 convs; half-waves split channels ----
  for (int c = hwb; c < C_; c += 2) {
    float n[3][3];
    #pragma unroll
    for (int dy = -1; dy <= 1; ++dy) {
      int hh = (h + dy + H_) & (H_ - 1);
      #pragma unroll
      for (int dx = -1; dx <= 1; ++dx) {
        int ww = (wcol + dx + W_) & (W_ - 1);
        n[dy + 1][dx + 1] = getv(c, hh, ww);
      }
    }
    float ctr = n[1][1];
    float sx = -n[0][0] + n[0][2] - 2.0f * n[1][0] + 2.0f * n[1][2] - n[2][0] + n[2][2];
    float sy = -n[0][0] - 2.0f * n[0][1] - n[0][2] + n[2][0] + 2.0f * n[2][1] + n[2][2];
    float lp = n[0][0] + 2.0f * n[0][1] + n[0][2] + 2.0f * n[1][0] - 12.0f * ctr
             + 2.0f * n[1][2] + n[2][0] + 2.0f * n[2][1] + n[2][2];
    featL[wv][p][c]              = (_Float16)ctr;
    featL[wv][p][C_ + 3 * c + 0] = (_Float16)sx;
    featL[wv][p][C_ + 3 * c + 1] = (_Float16)sy;
    featL[wv][p][C_ + 3 * c + 2] = (_Float16)lp;
  }
  __syncthreads();

  // ---- WMMA: hidden = w1 * feat   (6 M-tiles x 3 K-steps) ----
  v8f acc[6];
  #pragma unroll
  for (int mt = 0; mt < 6; ++mt)
    #pragma unroll
    for (int r = 0; r < 8; ++r) acc[mt][r] = 0.0f;

  const int off1 = hwb * 8;
  const int off2 = off1 + 16;
  #pragma unroll
  for (int k = 0; k < 3; ++k) {
    v8h lo = *(const v8h*)&featL[wv][p][32 * k + off1];
    v8h hi = *(const v8h*)&featL[wv][p][32 * k + off2];
    v16h bm = __builtin_shufflevector(lo, hi, 0, 1, 2, 3, 4, 5, 6, 7,
                                      8, 9, 10, 11, 12, 13, 14, 15);
    #pragma unroll
    for (int mt = 0; mt < 6; ++mt) {
      v16h am = *(const v16h*)&w1L[mt * 3 + k][lane][0];
      acc[mt] = __builtin_amdgcn_wmma_f32_16x16x32_f16(
          false, am, false, bm, (short)0, acc[mt], false, false);
    }
  }

  // ---- bias + ReLU + 96->3 layer; rows M=r (lanes<16) / M=r+8 (lanes>=16) ----
  float part[3] = {0.0f, 0.0f, 0.0f};
  const int ro = hwb * 8;
  #pragma unroll
  for (int mt = 0; mt < 6; ++mt) {
    #pragma unroll
    for (int r = 0; r < 8; ++r) {
      int o = mt * 16 + r + ro;
      float hv = fmaxf(acc[mt][r] + b1L[o], 0.0f);
      part[0] += w2L[0][o] * hv;
      part[1] += w2L[1][o] * hv;
      part[2] += w2L[2][o] * hv;
    }
  }
  #pragma unroll
  for (int j = 0; j < 3; ++j)
    part[j] += __shfl_xor(part[j], 16, 32);

  // ---- masked gene update into channels 20..22 ----
  if (lane < 16) {
    size_t pix = (size_t)bi * HW_ + (size_t)h * W_ + wcol;
    float um = floorf(noise[pix] + 0.5f);
    float mx = -3.0e38f;
    #pragma unroll
    for (int dy = -1; dy <= 1; ++dy) {
      int hh = (h + dy + H_) & (H_ - 1);
      #pragma unroll
      for (int dx = -1; dx <= 1; ++dx) {
        int ww = (wcol + dx + W_) & (W_ - 1);
        mx = fmaxf(mx, xin[((size_t)(bi * C_ + 3) * H_ + hh) * W_ + ww]);
      }
    }
    float s = um * ((mx > 0.1f) ? 1.0f : 0.0f);
    #pragma unroll
    for (int j = 0; j < 3; ++j) {
      float gbase = xg[((size_t)(bi * C_ + 13 + j) * H_ + h) * W_ + wcol];
      xg[((size_t)(bi * C_ + 20 + j) * H_ + h) * W_ + wcol] = gbase + part[j] * s;
    }
  }
}

// ---------------------------------------------------------------------------
extern "C" void kernel_launch(void* const* d_in, const int* in_sizes, int n_in,
                              void* d_out, int out_size, void* d_ws, size_t ws_size,
                              hipStream_t stream) {
  const float* x     = (const float*)d_in[0];
  const float* noise = (const float*)d_in[1];
  const float* w1w   = (const float*)d_in[2];
  const float* w1b   = (const float*)d_in[3];
  const float* w2w   = (const float*)d_in[4];
  const float* sw    = (const float*)d_in[5];
  const float* sb    = (const float*)d_in[6];
  const float* mw    = (const float*)d_in[7];
  const float* mb    = (const float*)d_in[8];
  const float* pAl   = (const float*)d_in[9];
  const float* pBe   = (const float*)d_in[10];
  const float* pOm   = (const float*)d_in[11];
  const float* pK    = (const float*)d_in[12];
  const float* pKp   = (const float*)d_in[13];

  float* out   = (float*)d_out;
  float* xg    = out;                                  // (B,C,H,W)
  float* phase = out + (size_t)B_ * C_ * HW_;          // (B,1,H,W)
  float* amp   = phase + (size_t)B_ * HW_;             // (B,1,H,W)

  float* na = (float*)d_ws;                            // (B,H,W)
  float* nb = na + (size_t)B_ * HW_;                   // (B,H,W)

  const int npix = B_ * HW_;
  geneca_osc1<<<npix / 256, 256, 0, stream>>>(x, sw, sb, pAl, pBe, pOm, pK, pKp,
                                              xg, na, nb);
  geneca_osc2<<<npix / 256, 256, 0, stream>>>(na, nb, xg, phase, amp);
  geneca_mlp<<<B_ * H_ * 4, 128, 0, stream>>>(x, xg, noise, w1w, w1b, w2w, mw, mb);
}
